// FalconH1ForCausalLM_46273977647124
// MI455X (gfx1250) — matrix operand order repaired and checked by reading, not compiled, not run on hardware
//
#include <hip/hip_runtime.h>

typedef __attribute__((ext_vector_type(16))) _Float16 v16h;
typedef __attribute__((ext_vector_type(8)))  _Float16 v8h;
typedef __attribute__((ext_vector_type(8)))  float    v8f;

namespace cfg {
constexpr int T = 2048, HID = 2048;
constexpr int NH = 32, NKV = 8, HD = 64;
constexpr int DSSM = 4096, DSTATE = 128, MH = 64;
constexpr int FFI = 8192;
constexpr int QKV_N = (NH + 2 * NKV) * HD;          // 3072
constexpr int ZXB_N = 2 * DSSM + 2 * DSTATE + MH;   // 8512
constexpr int XBC_N = DSSM + 2 * DSTATE;            // 4352
constexpr float EPS = 1e-5f;
constexpr float ATTN_IN = 1.1f, ATTN_OUT = 0.9f, SSM_IN = 1.2f, SSM_OUT = 0.8f;
constexpr float KEY_MULT = 0.7f, GATE_MULT = 1.3f, DOWN_MULT = 0.6f;
}

// ---------------------------------------------------------------------------
// helpers
// ---------------------------------------------------------------------------
__device__ inline v16h cat16(v8h lo, v8h hi) {
  v16h r;
#pragma unroll
  for (int i = 0; i < 8; ++i) { r[i] = lo[i]; r[8 + i] = hi[i]; }
  return r;
}

__device__ inline float block_reduce_sum_256(float v, float* sred) {
#pragma unroll
  for (int off = 16; off; off >>= 1) v += __shfl_down(v, off, 32);
  int wid = threadIdx.x >> 5, lane = threadIdx.x & 31;
  if (lane == 0) sred[wid] = v;
  __syncthreads();
  float tot = 0.f;
  if (threadIdx.x == 0) {
#pragma unroll
    for (int i = 0; i < 8; ++i) tot += sred[i];
    sred[0] = tot;
  }
  __syncthreads();
  tot = sred[0];
  __syncthreads();
  return tot;
}

// ---------------------------------------------------------------------------
// GEMM: C[M,N] = alpha * A(f16)[M,K] . B(f32)[N,K]^T  (+ Cadd)
// block = 256 threads (8 waves), tile 64(M) x 128(N), K-step 32.
// Double-buffered LDS for the fp32->f16 B tile; WMMA f32_16x16x32_f16.
// Interior blocks take a guard-free path; global_prefetch pulls B ahead.
// ---------------------------------------------------------------------------
#define BS_STRIDE 40

__global__ __launch_bounds__(256) void gemm_f16w_kernel(
    const _Float16* __restrict__ A, const float* __restrict__ B,
    float* __restrict__ C, const float* __restrict__ Cadd,
    int M, int N, int K, float alpha) {
  __shared__ _Float16 Bs[2][128 * BS_STRIDE];
  const int tid = threadIdx.x;
  const int wave = tid >> 5, lane = tid & 31;
  const int hf = lane >> 4, col = lane & 15;
  const int wm = wave & 3, wn = wave >> 2;  // 4 waves over M, 2 over N
  const int m0 = blockIdx.y * 64 + wm * 16;
  const int nblock = blockIdx.x * 128;
  const bool interior = (nblock + 128 <= N);  // block-uniform

  v8f acc[4] = {v8f{}, v8f{}, v8f{}, v8f{}};
  const _Float16* Arow = A + (size_t)(m0 + col) * K;

  float4 breg[4];
  auto loadB = [&](int k0) {
#pragma unroll
    for (int j = 0; j < 4; ++j) {
      int idx = tid + j * 256;      // 0..1023
      int row = idx >> 3;           // 0..127
      int kc = (idx & 7) * 4;       // 0..28
      int n = nblock + row;
      if (interior || n < N)
        breg[j] = *(const float4*)(B + (size_t)n * K + k0 + kc);
      else
        breg[j] = make_float4(0.f, 0.f, 0.f, 0.f);
    }
  };
  auto storeB = [&](int buf) {
#pragma unroll
    for (int j = 0; j < 4; ++j) {
      int idx = tid + j * 256;
      int row = idx >> 3;
      int kc = (idx & 7) * 4;
      _Float16* dst = &Bs[buf][row * BS_STRIDE + kc];
      dst[0] = (_Float16)breg[j].x; dst[1] = (_Float16)breg[j].y;
      dst[2] = (_Float16)breg[j].z; dst[3] = (_Float16)breg[j].w;
    }
  };

  loadB(0);
  storeB(0);
  __syncthreads();
  int cur = 0;

  const int prow = tid & 127;  // one prefetch row per thread
  for (int k0 = 0; k0 < K; k0 += 32) {
    const int knext = k0 + 32;
    if (knext < K) loadB(knext);  // issue global loads before the math
    if (interior && k0 + 64 < K)
      __builtin_prefetch(B + (size_t)(nblock + prow) * K + k0 + 64, 0, 1);

    const int kb = k0 + (hf ? 8 : 0);
    v16h a = cat16(*(const v8h*)(Arow + kb), *(const v8h*)(Arow + kb + 16));
#pragma unroll
    for (int t = 0; t < 4; ++t) {
      const _Float16* bp = &Bs[cur][(wn * 64 + t * 16 + col) * BS_STRIDE + (hf ? 8 : 0)];
      v16h b = cat16(*(const v8h*)bp, *(const v8h*)(bp + 16));
      acc[t] = __builtin_amdgcn_wmma_f32_16x16x32_f16(
          false, a, false, b, (short)0, acc[t], false, false);
    }
    if (knext < K) {
      storeB(1 - cur);   // writes go to the other buffer: no read conflict
      __syncthreads();   // single barrier per K-step
      cur ^= 1;
    }
  }

#pragma unroll
  for (int t = 0; t < 4; ++t) {
    int n = nblock + wn * 64 + t * 16 + col;
    if (n >= N) continue;
#pragma unroll
    for (int r = 0; r < 8; ++r) {
      int m = m0 + r + hf * 8;
      size_t off = (size_t)m * N + n;
      float v = acc[t][r] * alpha;
      if (Cadd) v += Cadd[off];
      C[off] = v;
    }
  }
}

// ---------------------------------------------------------------------------
// RMSNorm (f32 in) -> f16 out
// ---------------------------------------------------------------------------
__global__ __launch_bounds__(256) void rmsnorm16_kernel(
    const float* __restrict__ x, const float* __restrict__ w,
    _Float16* __restrict__ out, int D) {
  __shared__ float sred[8];
  const int t = blockIdx.x;
  const float* row = x + (size_t)t * D;
  float ss = 0.f;
  for (int c = threadIdx.x; c < D; c += 256) { float v = row[c]; ss += v * v; }
  float tot = block_reduce_sum_256(ss, sred);
  float rs = rsqrtf(tot / (float)D + cfg::EPS);
  for (int c = threadIdx.x; c < D; c += 256)
    out[(size_t)t * D + c] = (_Float16)(row[c] * rs * w[c]);
}

// ---------------------------------------------------------------------------
// RoPE + pack q/k(f16) + transpose v -> vT (f16 [NKV*HD, T])
// ---------------------------------------------------------------------------
__global__ __launch_bounds__(256) void rope_pack_kernel(
    const float* __restrict__ qkv, const int* __restrict__ positions,
    _Float16* __restrict__ q16, _Float16* __restrict__ k16,
    _Float16* __restrict__ vT16) {
  using namespace cfg;
  const int t = blockIdx.x;
  const float pos = (float)positions[t];
  const float LOGB = 25.328436022934504f;  // ln(1e11)
  const float* row = qkv + (size_t)t * QKV_N;

  for (int idx = threadIdx.x; idx < NH * 32; idx += 256) {
    int h = idx >> 5, i = idx & 31;
    float inv = __expf(-(2.f * i / (float)HD) * LOGB);
    float fr = pos * inv, c = __cosf(fr), s = __sinf(fr);
    float x1 = row[h * HD + i], x2 = row[h * HD + i + 32];
    _Float16* q = q16 + (size_t)t * (NH * HD) + h * HD;
    q[i] = (_Float16)(x1 * c - x2 * s);
    q[i + 32] = (_Float16)(x1 * s + x2 * c);
  }
  for (int idx = threadIdx.x; idx < NKV * 32; idx += 256) {
    int h = idx >> 5, i = idx & 31;
    float inv = __expf(-(2.f * i / (float)HD) * LOGB);
    float fr = pos * inv, c = __cosf(fr), s = __sinf(fr);
    const float* kr = row + NH * HD + h * HD;
    float x1 = kr[i] * KEY_MULT, x2 = kr[i + 32] * KEY_MULT;
    _Float16* k = k16 + (size_t)t * (NKV * HD) + h * HD;
    k[i] = (_Float16)(x1 * c - x2 * s);
    k[i + 32] = (_Float16)(x1 * s + x2 * c);
  }
  for (int idx = threadIdx.x; idx < NKV * HD; idx += 256) {
    vT16[(size_t)idx * T + t] = (_Float16)row[(NH + NKV) * HD + idx];
  }
}

// ---------------------------------------------------------------------------
// Flash attention: one wave per (16 query rows, head). WMMA for Q.K^T and P.V
// ---------------------------------------------------------------------------
__global__ __launch_bounds__(32) void flash_attn_kernel(
    const _Float16* __restrict__ q16, const _Float16* __restrict__ k16,
    const _Float16* __restrict__ vT16, _Float16* __restrict__ attn16) {
  using namespace cfg;
  __shared__ _Float16 pbuf[16][32];
  const int m0 = blockIdx.x * 16;
  const int h = blockIdx.y;
  const int hk = h >> 2;  // NH/NKV = 4
  const int lane = threadIdx.x;
  const int hf = lane >> 4, col = lane & 15;
  const float scale = 0.125f;  // HD^-0.5

  const _Float16* qrow = q16 + (size_t)(m0 + col) * (NH * HD) + h * HD;
  v16h aq[2];
#pragma unroll
  for (int ks = 0; ks < 2; ++ks) {
    int kb = ks * 32 + (hf ? 8 : 0);
    aq[ks] = cat16(*(const v8h*)(qrow + kb), *(const v8h*)(qrow + kb + 16));
  }

  v8f o[4] = {v8f{}, v8f{}, v8f{}, v8f{}};
  float mstate[8], lstate[8];
#pragma unroll
  for (int r = 0; r < 8; ++r) { mstate[r] = -1e30f; lstate[r] = 0.f; }

  for (int n0 = 0; n0 < m0 + 16; n0 += 32) {
    v8f s[2] = {v8f{}, v8f{}};
#pragma unroll
    for (int tile = 0; tile < 2; ++tile) {
#pragma unroll
      for (int ks = 0; ks < 2; ++ks) {
        const _Float16* kp = k16 + (size_t)(n0 + tile * 16 + col) * (NKV * HD) +
                             hk * HD + ks * 32 + (hf ? 8 : 0);
        v16h b = cat16(*(const v8h*)kp, *(const v8h*)(kp + 16));
        s[tile] = __builtin_amdgcn_wmma_f32_16x16x32_f16(
            false, aq[ks], false, b, (short)0, s[tile], false, false);
      }
    }
    float p0[8], p1[8];
#pragma unroll
    for (int r = 0; r < 8; ++r) {
      int m = m0 + r + hf * 8;
      float s0 = s[0][r] * scale, s1 = s[1][r] * scale;
      if (n0 + col > m) s0 = -1e30f;
      if (n0 + 16 + col > m) s1 = -1e30f;
      float mx = fmaxf(s0, s1);
#pragma unroll
      for (int off = 1; off < 16; off <<= 1) mx = fmaxf(mx, __shfl_xor(mx, off, 32));
      float mnew = fmaxf(mstate[r], mx);
      float corr = __expf(mstate[r] - mnew);
      p0[r] = __expf(s0 - mnew);
      p1[r] = __expf(s1 - mnew);
      float rs = p0[r] + p1[r];
#pragma unroll
      for (int off = 1; off < 16; off <<= 1) rs += __shfl_xor(rs, off, 32);
      lstate[r] = lstate[r] * corr + rs;
      mstate[r] = mnew;
#pragma unroll
      for (int tt = 0; tt < 4; ++tt) o[tt][r] *= corr;
    }
    __syncthreads();
#pragma unroll
    for (int r = 0; r < 8; ++r) {
      pbuf[r + hf * 8][col] = (_Float16)p0[r];
      pbuf[r + hf * 8][16 + col] = (_Float16)p1[r];
    }
    __syncthreads();
    v16h ap;
    {
      int kb = hf ? 8 : 0;
      const _Float16* pp = &pbuf[col][0];
      ap = cat16(*(const v8h*)(pp + kb), *(const v8h*)(pp + kb + 16));
    }
#pragma unroll
    for (int tt = 0; tt < 4; ++tt) {
      const _Float16* vp = vT16 + (size_t)(hk * HD + tt * 16 + col) * T + n0 + (hf ? 8 : 0);
      v16h b = cat16(*(const v8h*)vp, *(const v8h*)(vp + 16));
      o[tt] = __builtin_amdgcn_wmma_f32_16x16x32_f16(
          false, ap, false, b, (short)0, o[tt], false, false);
    }
    __syncthreads();
  }
#pragma unroll
  for (int r = 0; r < 8; ++r) {
    float inv = 1.f / lstate[r];
    _Float16* orow = attn16 + (size_t)(m0 + r + hf * 8) * (NH * HD) + h * HD;
#pragma unroll
    for (int tt = 0; tt < 4; ++tt)
      orow[tt * 16 + col] = (_Float16)(o[tt][r] * inv);
  }
}

// ---------------------------------------------------------------------------
// Causal depthwise conv (K=4) + SiLU, with mup column scales on xBC
// ---------------------------------------------------------------------------
__global__ __launch_bounds__(256) void conv_silu_kernel(
    const float* __restrict__ zxbcdt, const float* __restrict__ conv_w,
    const float* __restrict__ conv_b, float* __restrict__ xBC) {
  using namespace cfg;
  const int c = blockIdx.x * 256 + threadIdx.x;  // < 4352
  const int t = blockIdx.y;
  const float mup = (c < DSSM) ? 0.9f : (c < DSSM + DSTATE ? 0.8f : 1.1f);
  float acc = conv_b[c];
#pragma unroll
  for (int kk = 0; kk < 4; ++kk) {
    int tt = t - 3 + kk;
    if (tt >= 0)
      acc += zxbcdt[(size_t)tt * ZXB_N + DSSM + c] * mup * conv_w[c * 4 + kk];
  }
  xBC[(size_t)t * XBC_N + c] = acc / (1.f + __expf(-acc));  // silu
}

// ---------------------------------------------------------------------------
// dt: softplus(dt*1.2 + bias), dA = exp(dtp * -exp(A_log))
// ---------------------------------------------------------------------------
__global__ __launch_bounds__(64) void dt_kernel(
    const float* __restrict__ zxbcdt, const float* __restrict__ dt_bias,
    const float* __restrict__ A_log, float* __restrict__ dtp,
    float* __restrict__ dA) {
  using namespace cfg;
  const int t = blockIdx.x, h = threadIdx.x;
  float raw = zxbcdt[(size_t)t * ZXB_N + 2 * DSSM + 2 * DSTATE + h] * 1.2f + dt_bias[h];
  float sp = (raw > 20.f) ? raw : log1pf(__expf(raw));
  dtp[t * MH + h] = sp;
  dA[t * MH + h] = __expf(sp * (-__expf(A_log[h])));
}

// ---------------------------------------------------------------------------
// SSM scan: one WAVE per (h,p); lane owns 4 of the 128 states (float4 B/C).
// Pure shfl reduction: zero barriers across the 2048-step recurrence.
// ---------------------------------------------------------------------------
__global__ __launch_bounds__(32) void ssm_scan_kernel(
    const float* __restrict__ xBC, const float* __restrict__ dtp,
    const float* __restrict__ dA, const float* __restrict__ Dv,
    float* __restrict__ y) {
  using namespace cfg;
  const int h = blockIdx.x, p = blockIdx.y;
  const int lane = threadIdx.x;
  const float dcoef = Dv[h];
  float st0 = 0.f, st1 = 0.f, st2 = 0.f, st3 = 0.f;
  for (int t = 0; t < T; ++t) {
    const float* row = xBC + (size_t)t * XBC_N;
    float dAv = dA[t * MH + h];
    float dtv = dtp[t * MH + h];
    float xv = row[h * 64 + p];
    float4 Bv = *(const float4*)(row + DSSM + lane * 4);
    float4 Cv = *(const float4*)(row + DSSM + DSTATE + lane * 4);
    float coef = dtv * xv;
    st0 = st0 * dAv + coef * Bv.x;
    st1 = st1 * dAv + coef * Bv.y;
    st2 = st2 * dAv + coef * Bv.z;
    st3 = st3 * dAv + coef * Bv.w;
    float contrib = st0 * Cv.x + st1 * Cv.y + st2 * Cv.z + st3 * Cv.w;
#pragma unroll
    for (int off = 16; off; off >>= 1) contrib += __shfl_down(contrib, off, 32);
    if (lane == 0) y[(size_t)t * DSSM + h * 64 + p] = contrib + dcoef * xv;
  }
}

// ---------------------------------------------------------------------------
// gated RMSNorm: rmsnorm(y * silu(z), w) -> f16
// ---------------------------------------------------------------------------
__global__ __launch_bounds__(256) void gatednorm_kernel(
    const float* __restrict__ y, const float* __restrict__ zxbcdt,
    const float* __restrict__ w, _Float16* __restrict__ out) {
  using namespace cfg;
  __shared__ float sred[8];
  const int t = blockIdx.x;
  const float* yr = y + (size_t)t * DSSM;
  const float* zr = zxbcdt + (size_t)t * ZXB_N;  // z cols, mup=1.0
  float ss = 0.f;
  for (int c = threadIdx.x; c < DSSM; c += 256) {
    float z = zr[c];
    float v = yr[c] * (z / (1.f + __expf(-z)));
    ss += v * v;
  }
  float tot = block_reduce_sum_256(ss, sred);
  float rs = rsqrtf(tot / (float)DSSM + EPS);
  for (int c = threadIdx.x; c < DSSM; c += 256) {
    float z = zr[c];
    float v = yr[c] * (z / (1.f + __expf(-z)));
    out[(size_t)t * DSSM + c] = (_Float16)(v * rs * w[c]);
  }
}

// ---------------------------------------------------------------------------
// SwiGLU: silu(g*1.3) * u -> f16
// ---------------------------------------------------------------------------
__global__ __launch_bounds__(256) void swiglu_kernel(
    const float* __restrict__ gu, _Float16* __restrict__ act16) {
  using namespace cfg;
  const int c = blockIdx.x * 256 + threadIdx.x;  // < FFI
  const int t = blockIdx.y;
  float g = gu[(size_t)t * (2 * FFI) + c] * GATE_MULT;
  float u = gu[(size_t)t * (2 * FFI) + FFI + c];
  act16[(size_t)t * FFI + c] = (_Float16)((g / (1.f + __expf(-g))) * u);
}

// ---------------------------------------------------------------------------
// host orchestration
// ---------------------------------------------------------------------------
extern "C" void kernel_launch(void* const* d_in, const int* in_sizes, int n_in,
                              void* d_out, int out_size, void* d_ws, size_t ws_size,
                              hipStream_t stream) {
  using namespace cfg;
  const int* positions = (const int*)d_in[0];
  const float* hidden = (const float*)d_in[1];
  const float* input_ln_w = (const float*)d_in[2];
  const float* qkv_w = (const float*)d_in[3];
  const float* o_w = (const float*)d_in[4];
  const float* in_proj_w = (const float*)d_in[5];
  const float* conv_w = (const float*)d_in[6];
  const float* conv_b = (const float*)d_in[7];
  const float* A_log = (const float*)d_in[8];
  const float* Dv = (const float*)d_in[9];
  const float* dt_bias = (const float*)d_in[10];
  const float* ssm_norm_w = (const float*)d_in[11];
  const float* out_proj_w = (const float*)d_in[12];
  const float* pre_ff_ln_w = (const float*)d_in[13];
  const float* gate_up_w = (const float*)d_in[14];
  const float* down_w = (const float*)d_in[15];
  float* out = (float*)d_out;

  char* p = (char*)d_ws;
  auto alloc = [&](size_t bytes) -> char* {
    char* r = p;
    p += (bytes + 255) & ~(size_t)255;
    return r;
  };
  _Float16* hs16 = (_Float16*)alloc((size_t)T * HID * 2);
  float* qkv = (float*)alloc((size_t)T * QKV_N * 4);
  _Float16* q16 = (_Float16*)alloc((size_t)T * NH * HD * 2);
  _Float16* k16 = (_Float16*)alloc((size_t)T * NKV * HD * 2);
  _Float16* vT16 = (_Float16*)alloc((size_t)NKV * HD * T * 2);
  _Float16* attn16 = (_Float16*)alloc((size_t)T * NH * HD * 2);
  float* hid_part = (float*)alloc((size_t)T * HID * 4);
  float* zxbcdt = (float*)alloc((size_t)T * ZXB_N * 4);
  float* xBC = (float*)alloc((size_t)T * XBC_N * 4);
  float* dtp = (float*)alloc((size_t)T * MH * 4);
  float* dA = (float*)alloc((size_t)T * MH * 4);
  float* yssm = (float*)alloc((size_t)T * DSSM * 4);
  _Float16* y16 = (_Float16*)alloc((size_t)T * DSSM * 2);
  float* hidden2 = (float*)alloc((size_t)T * HID * 4);
  _Float16* h16 = (_Float16*)alloc((size_t)T * HID * 2);
  float* gu = (float*)alloc((size_t)T * 2 * FFI * 4);
  _Float16* act16 = (_Float16*)alloc((size_t)T * FFI * 2);

  rmsnorm16_kernel<<<T, 256, 0, stream>>>(hidden, input_ln_w, hs16, HID);
  gemm_f16w_kernel<<<dim3(QKV_N / 128, T / 64), 256, 0, stream>>>(
      hs16, qkv_w, qkv, nullptr, T, QKV_N, HID, ATTN_IN);
  rope_pack_kernel<<<T, 256, 0, stream>>>(qkv, positions, q16, k16, vT16);
  flash_attn_kernel<<<dim3(T / 16, NH), 32, 0, stream>>>(q16, k16, vT16, attn16);
  gemm_f16w_kernel<<<dim3(HID / 128, T / 64), 256, 0, stream>>>(
      attn16, o_w, hid_part, hidden, T, HID, NH * HD, ATTN_OUT);
  gemm_f16w_kernel<<<dim3((ZXB_N + 127) / 128, T / 64), 256, 0, stream>>>(
      hs16, in_proj_w, zxbcdt, nullptr, T, ZXB_N, HID, SSM_IN);
  conv_silu_kernel<<<dim3(XBC_N / 256, T), 256, 0, stream>>>(zxbcdt, conv_w, conv_b, xBC);
  dt_kernel<<<T, 64, 0, stream>>>(zxbcdt, dt_bias, A_log, dtp, dA);
  ssm_scan_kernel<<<dim3(MH, 64), 32, 0, stream>>>(xBC, dtp, dA, Dv, yssm);
  gatednorm_kernel<<<T, 256, 0, stream>>>(yssm, zxbcdt, ssm_norm_w, y16);
  gemm_f16w_kernel<<<dim3(HID / 128, T / 64), 256, 0, stream>>>(
      y16, out_proj_w, hidden2, hid_part, T, HID, DSSM, SSM_OUT);
  rmsnorm16_kernel<<<T, 256, 0, stream>>>(hidden2, pre_ff_ln_w, h16, HID);
  gemm_f16w_kernel<<<dim3(2 * FFI / 128, T / 64), 256, 0, stream>>>(
      h16, gate_up_w, gu, nullptr, T, 2 * FFI, HID, 1.0f);
  swiglu_kernel<<<dim3(FFI / 256, T), 256, 0, stream>>>(gu, act16);
  gemm_f16w_kernel<<<dim3(HID / 128, T / 64), 256, 0, stream>>>(
      act16, down_w, out, hidden2, T, HID, FFI, DOWN_MULT);
}
